// GAT_64493228917147
// MI455X (gfx1250) — matrix operand order repaired and checked by reading, not compiled, not run on hardware
//
#include <hip/hip_runtime.h>
#include <hip/hip_bf16.h>

// ---------------------------------------------------------------------------
// Fused GAT layer for MI455X (gfx1250, wave32, WMMA).
//   x:[4,50000,10,128] f32, W:[128,64] f32, a:[4,16] f32 -> out:[4,64,50000] f32
// GEMM path: x rows converted to f16 in-register, W pre-swizzled to f16 in LDS,
// projection via v_wmma_f32_16x16x32_f16 (f32 accumulate). One node (K=10
// neighbors padded to a 16-row M-tile) per wave per pass.
// ---------------------------------------------------------------------------

typedef __attribute__((ext_vector_type(16))) _Float16 v16h;
typedef __attribute__((ext_vector_type(8)))  float    v8f;
typedef __attribute__((ext_vector_type(4)))  float    f32x4;

#define GAT_B   4
#define GAT_N   50000
#define GAT_K   10
#define GAT_F   128
#define GAT_C   64      // 4*H output channels
#define GAT_H   16      // heads

#define WAVES   8
#define BLOCK   (WAVES * 32)

__global__ __launch_bounds__(BLOCK)
void gat_wmma_kernel(const float* __restrict__ x,
                     const float* __restrict__ W,
                     const float* __restrict__ a,
                     float* __restrict__ out,
                     int n_groups)
{
    // W in f16, pre-swizzled into WMMA B-operand layout: [nt][ks][lane][16 halfs]
    __shared__ _Float16 bsw[4 * 4 * 32 * 16];        // 16 KB
    // per-wave hw tile (C = A*B), rows = k neighbors, cols = 4h+d
    __shared__ float    hw_s[WAVES][16][GAT_C];      // 32 KB
    // output staging for coalesced transposed stores
    __shared__ float    outs[GAT_C][WAVES];          //  2 KB

    const int tid  = threadIdx.x;
    const int wave = tid >> 5;
    const int lane = tid & 31;
    const int lo   = lane & 15;
    const int hi   = lane >> 4;           // which K half-block this lane owns

    // ---- build swizzled B operand (done once per block) ----
    // lane l<16 : halfs j hold W[ks*32     + j][nt*16 + l]
    // lane l>=16: halfs j hold W[ks*32 + 16 + j][nt*16 + (l-16)]
    for (int e = tid; e < 4 * 4 * 32 * 16; e += BLOCK) {
        const int j  = e & 15;
        const int l  = (e >> 4) & 31;
        const int ks = (e >> 9) & 3;
        const int nt = e >> 11;
        const int kk = ks * 32 + ((l >> 4) << 4) + j;
        const int nc = nt * 16 + (l & 15);
        bsw[e] = (_Float16)W[kk * GAT_C + nc];
    }

    // per-lane attention vector a[d][h] (lanes 0..15 use h = lane)
    const float a0 = a[0 * GAT_H + lo];
    const float a1 = a[1 * GAT_H + lo];
    const float a2 = a[2 * GAT_H + lo];
    const float a3 = a[3 * GAT_H + lo];

    __syncthreads();

    // row of the padded 16-row A tile this lane loads (rows 10..15 clamp to 0:
    // C rows are independent and rows >= 10 are never read back)
    const int rr = (lo < GAT_K) ? lo : 0;

    for (int g = blockIdx.x; g < n_groups; g += gridDim.x) {
        const int item = g * WAVES + wave;               // flat b*N + n
        const float* rowp = x + ((long)item * GAT_K + rr) * GAT_F;

        // ---- A fragments: 16-bit A 16x32, contiguous k order (matches bsw) ----
        // lane<16 : halfs 0..15 = K o0..o0+15
        // lane>=16: same with o0 += 16
        v16h afrag[4];
        #pragma unroll
        for (int ks = 0; ks < 4; ++ks) {
            const int o0 = ks * 32 + hi * 16;
            f32x4 f0 = *(const f32x4*)(rowp + o0);
            f32x4 f1 = *(const f32x4*)(rowp + o0 + 4);
            f32x4 f2 = *(const f32x4*)(rowp + o0 + 8);
            f32x4 f3 = *(const f32x4*)(rowp + o0 + 12);
            union { v16h v; _Float16 h[16]; } u;
            #pragma unroll
            for (int j = 0; j < 4; ++j) {
                u.h[j]      = (_Float16)f0[j];
                u.h[4 + j]  = (_Float16)f1[j];
                u.h[8 + j]  = (_Float16)f2[j];
                u.h[12 + j] = (_Float16)f3[j];
            }
            afrag[ks] = u.v;
        }

        // ---- 16 WMMAs: 4 N-tiles x 4 K-steps, f32 accumulate ----
        const v8f zero8 = {0.f, 0.f, 0.f, 0.f, 0.f, 0.f, 0.f, 0.f};
        v8f acc[4];
        #pragma unroll
        for (int nt = 0; nt < 4; ++nt) acc[nt] = zero8;

        #pragma unroll
        for (int nt = 0; nt < 4; ++nt) {
            #pragma unroll
            for (int ks = 0; ks < 4; ++ks) {
                const v16h bfrag =
                    *(const v16h*)&bsw[(((nt << 2) | ks) << 9) + (lane << 4)];
                acc[nt] = __builtin_amdgcn_wmma_f32_16x16x32_f16(
                    false, afrag[ks], false, bfrag,
                    (short)0, acc[nt], false, false);
            }
        }

        // ---- spill C tile to per-wave LDS (transpose for softmax over rows) ----
        // C layout: VGPR j, lane<16 -> (M=j, N=lo); lane>=16 -> (M=j+8, N=lo)
        float* hwp = &hw_s[wave][0][0];
        #pragma unroll
        for (int nt = 0; nt < 4; ++nt) {
            #pragma unroll
            for (int j = 0; j < 8; ++j) {
                hwp[(hi * 8 + j) * GAT_C + nt * 16 + lo] = acc[nt][j];
            }
        }

        // ---- per-head softmax over K neighbors + weighted sum (lanes 0..15) ----
        if (lane < GAT_H) {
            const int h4 = lane * 4;
            f32x4 hv[GAT_K];
            float lg[GAT_K];
            float m = -3.0e38f;
            #pragma unroll
            for (int k = 0; k < GAT_K; ++k) {
                hv[k] = *(const f32x4*)&hw_s[wave][k][h4];
                lg[k] = hv[k][0] * a0 + hv[k][1] * a1 +
                        hv[k][2] * a2 + hv[k][3] * a3;
                m = fmaxf(m, lg[k]);
            }
            float s = 0.f;
            #pragma unroll
            for (int k = 0; k < GAT_K; ++k) {
                lg[k] = __builtin_amdgcn_exp2f((lg[k] - m) * 1.44269504088896340736f);
                s += lg[k];
            }
            const float inv = __builtin_amdgcn_rcpf(s);
            f32x4 o = {0.f, 0.f, 0.f, 0.f};
            #pragma unroll
            for (int k = 0; k < GAT_K; ++k) {
                const float wk = lg[k] * inv;
                o += wk * hv[k];
            }
            #pragma unroll
            for (int j = 0; j < 4; ++j) {
                float v = o[j];
                v = fmaxf(v, 0.2f * v);        // leaky_relu(slope 0.2)
                outs[h4 + j][wave] = v;
            }
        }
        __syncthreads();

        // ---- coalesced transposed store: out[b][c][n], 8 consecutive n per c ----
        #pragma unroll
        for (int half = 0; half < 2; ++half) {
            const int e  = tid + half * BLOCK;     // 0..511
            const int c  = e >> 3;
            const int i  = e & 7;
            const int it = g * WAVES + i;
            const int bb = it / GAT_N;
            const int nn = it - bb * GAT_N;
            out[((long)bb * GAT_C + c) * GAT_N + nn] = outs[c][i];
        }
        __syncthreads();   // outs reused next pass
    }
}

extern "C" void kernel_launch(void* const* d_in, const int* in_sizes, int n_in,
                              void* d_out, int out_size, void* d_ws, size_t ws_size,
                              hipStream_t stream)
{
    (void)in_sizes; (void)n_in; (void)out_size; (void)d_ws; (void)ws_size;

    const float* x = (const float*)d_in[0];
    const float* W = (const float*)d_in[1];
    const float* a = (const float*)d_in[2];
    float* out = (float*)d_out;

    const int total    = GAT_B * GAT_N;        // 200000 nodes
    const int n_groups = total / WAVES;        // 25000 (exact)
    int grid = 2048;
    if (grid > n_groups) grid = n_groups;

    gat_wmma_kernel<<<dim3(grid), dim3(BLOCK), 0, stream>>>(x, W, a, out, n_groups);
}
